// MultiHeadAttention_56375740727430
// MI455X (gfx1250) — matrix operand order repaired and checked by reading, not compiled, not run on hardware
//
#include <hip/hip_runtime.h>
#include <hip/hip_bf16.h>
#include <stdint.h>

typedef __attribute__((ext_vector_type(8)))  float  v8f;
typedef __attribute__((ext_vector_type(16))) __bf16 v16bf;

#define WMMA_BF16(A, B, C) \
  __builtin_amdgcn_wmma_f32_16x16x32_bf16(false, (A), false, (B), (short)0, (C), false, false)

constexpr int Bc = 2, Sc = 2048, Dc = 1024, Hc = 16, KSc = 64;

// ---------------- TDM (Tensor Data Mover) support ---------------------------
#if defined(__AMDGCN__) && __has_builtin(__builtin_amdgcn_tensor_load_to_lds)
#define USE_TDM 1
#else
#define USE_TDM 0
#endif

#if USE_TDM
typedef unsigned int u32x4 __attribute__((ext_vector_type(4)));
typedef int          i32x4 __attribute__((ext_vector_type(4)));
typedef int          i32x8 __attribute__((ext_vector_type(8)));

__device__ __forceinline__ unsigned lds_addr_of(const void* p) {
  return (unsigned)(unsigned long long)(__attribute__((address_space(3))) const void*)p;
}

// 2-D bf16 tile load: tile_dim0 = 64 elements (128 B rows), row stride 64,
// LDS row padding +16 B (pad_interval: 32 DWORDs -> code 4; pad_amount: 4
// DWORDs -> code 3) to reproduce the 72-element padded LDS stride.
__device__ __forceinline__ void tdm_load_tile64(unsigned lds_off, const void* gptr,
                                                int rows) {
  const unsigned long long ga = (unsigned long long)gptr;
  u32x4 g0;
  g0[0] = 1u;                                            // count=1 (valid D#)
  g0[1] = lds_off;                                       // lds_addr
  g0[2] = (unsigned)ga;                                  // global_addr[31:0]
  g0[3] = (unsigned)((ga >> 32) & 0x01FFFFFFull)         // global_addr[56:32]
          | (2u << 30);                                  // type = 2 (image)
  i32x8 g1;
  g1[0] = (1 << 16)      // data_size = 1 (2 bytes)
        | (1 << 20)      // pad_enable
        | (4 << 22)      // pad_interval: 32 DWORDs
        | (3 << 25);     // pad_amount: 4 DWORDs (16 B)
  g1[1] = (64 << 16);    // tensor_dim0 = 64 (bits 79:48, low half)
  g1[2] = (2048 << 16);  // tensor_dim1 = 2048 (bits 111:80, low half)
  g1[3] = (64 << 16);    // tile_dim0 = 64 (bits 127:112)
  g1[4] = rows & 0xFFFF; // tile_dim1 = rows, tile_dim2 = 0
  g1[5] = 64;            // tensor_dim0_stride = 64 (low 32)
  g1[6] = 0;             // stride hi / dim1_stride lo
  g1[7] = 0;
  const i32x4 gz = {0, 0, 0, 0};
#if __clang_major__ >= 23
  const i32x8 gz8 = {0, 0, 0, 0, 0, 0, 0, 0};
  __builtin_amdgcn_tensor_load_to_lds(g0, g1, gz, gz, gz8, 0);
#else
  __builtin_amdgcn_tensor_load_to_lds(g0, g1, gz, gz, 0);
#endif
}
#endif  // USE_TDM

// ---------------- WMMA fragment loaders from LDS ---------------------------
// A-matrix 16x32 bf16 (per ISA): lanes 0-15 VGPR0-3 K=0..7, VGPR4-7 K=16..23;
// lanes 16-31 K=8..15 / 24..31.  Two ds_load_b128 per lane.
__device__ __forceinline__ v16bf frag_a(const __bf16* rowp, int half) {
  union { uint4 u[2]; v16bf v; } c;
  c.u[0] = *reinterpret_cast<const uint4*>(rowp + half * 8);
  c.u[1] = *reinterpret_cast<const uint4*>(rowp + 16 + half * 8);
  return c.v;
}
// B-matrix 32x16 bf16: lane = column N; lanes 0-15 hold K=0..7,16..23, lanes 16-31
// K=8..15,24..31.  B tiles are stored [n][k] in LDS, two ds_load_b128.
__device__ __forceinline__ v16bf frag_b(const __bf16* rowp, int half) {
  union { uint4 u[2]; v16bf v; } c;
  c.u[0] = *reinterpret_cast<const uint4*>(rowp + half * 8);
  c.u[1] = *reinterpret_cast<const uint4*>(rowp + 16 + half * 8);
  return c.v;
}

// ---- Kernel 1: QKV projection, per (b, h, 128-row s tile) ------------------
__global__ __launch_bounds__(256)
void proj_qkv_kernel(const float* __restrict__ x,
                     const float* __restrict__ Wq,
                     const float* __restrict__ Wk,
                     const float* __restrict__ Wv,
                     __bf16* __restrict__ Qb,
                     __bf16* __restrict__ Kb,
                     __bf16* __restrict__ Vb) {
  const int s0   = blockIdx.x * 128;
  const int h    = blockIdx.y;
  const int b    = blockIdx.z;
  const int tid  = threadIdx.x;
  const int w    = tid >> 5;
  const int lane = tid & 31;
  const int nIdx = lane & 15;
  const int halfSel = lane >> 4;

  __shared__ __align__(16) __bf16 lsX[128 * 40];     // x tile  [m][kk]
  __shared__ __align__(16) __bf16 lsW[3][64 * 40];   // W tiles [n][kk] (transposed)

  const float* Wm[3] = { Wq, Wk, Wv };

  v8f acc[3][4];
  const v8f vzero = {0.f,0.f,0.f,0.f,0.f,0.f,0.f,0.f};
#pragma unroll
  for (int m = 0; m < 3; ++m)
#pragma unroll
    for (int j = 0; j < 4; ++j) acc[m][j] = vzero;

  for (int d0 = 0; d0 < Dc; d0 += 32) {
    __syncthreads();
    {
      // x tile: 128 rows x 32 cols f32 -> bf16, packed ds_store_b128
      const int row = tid >> 1;
      const int cb  = (tid & 1) * 16;
      const float4* src = reinterpret_cast<const float4*>(
          x + ((size_t)b * Sc + s0 + row) * Dc + d0 + cb);
      float fv[16];
#pragma unroll
      for (int i = 0; i < 4; ++i) {
        float4 f = src[i];
        fv[4*i+0] = f.x; fv[4*i+1] = f.y; fv[4*i+2] = f.z; fv[4*i+3] = f.w;
      }
      union { __bf16 e[16]; uint4 u[2]; } pk;
#pragma unroll
      for (int i = 0; i < 16; ++i) pk.e[i] = (__bf16)fv[i];
      *reinterpret_cast<uint4*>(&lsX[row * 40 + cb])     = pk.u[0];
      *reinterpret_cast<uint4*>(&lsX[row * 40 + cb + 8]) = pk.u[1];

      // W tiles for this head, transposed into [n][kk]; 8 contiguous kk/thread
      const int n   = tid & 63;
      const int kkb = (tid >> 6) * 8;
#pragma unroll
      for (int m = 0; m < 3; ++m) {
        const float* Wp = Wm[m];
        union { __bf16 e[8]; uint4 u; } pw;
#pragma unroll
        for (int i = 0; i < 8; ++i)
          pw.e[i] = (__bf16)Wp[(size_t)(d0 + kkb + i) * (KSc * Hc) + n * Hc + h];
        *reinterpret_cast<uint4*>(&lsW[m][n * 40 + kkb]) = pw.u;
      }
    }
    __syncthreads();

    const v16bf aF = frag_a(&lsX[(16 * w + nIdx) * 40], halfSel);
#pragma unroll
    for (int m = 0; m < 3; ++m) {
      v16bf bF[4];
#pragma unroll
      for (int j = 0; j < 4; ++j)
        bF[j] = frag_b(&lsW[m][(16 * j + nIdx) * 40], halfSel);
#pragma unroll
      for (int j = 0; j < 4; ++j)
        acc[m][j] = WMMA_BF16(aF, bF[j], acc[m][j]);
    }
  }

  __bf16* outp[3] = { Qb, Kb, Vb };
  const size_t base = (((size_t)b * Hc + h) * Sc + s0 + 16 * w) * KSc;
#pragma unroll
  for (int m = 0; m < 3; ++m)
#pragma unroll
    for (int j = 0; j < 4; ++j)
#pragma unroll
      for (int r = 0; r < 8; ++r) {
        const int M = r + 8 * halfSel;  // C layout: VGPR r = rows r / r+8
        outp[m][base + (size_t)M * KSc + 16 * j + nIdx] = (__bf16)acc[m][j][r];
      }
}

// ---- Kernel 2: causal flash attention, per (b, h, 128-query tile) ----------
__global__ __launch_bounds__(256)
void attn_flash_kernel(const __bf16* __restrict__ Qb,
                       const __bf16* __restrict__ Kb,
                       const __bf16* __restrict__ Vb,
                       __bf16* __restrict__ Ob) {
  const int q0   = blockIdx.x * 128;
  const int h    = blockIdx.y;
  const int b    = blockIdx.z;
  const int tid  = threadIdx.x;
  const int w    = tid >> 5;
  const int lane = tid & 31;
  const int nIdx = lane & 15;
  const int halfSel = lane >> 4;

  __shared__ __align__(16) __bf16 lsQ[128 * 72];     // Q  [m][k]
  __shared__ __align__(16) __bf16 lsK[64 * 72];      // K  [l][k]
  __shared__ __align__(16) __bf16 lsVT[64 * 72];     // V^T[dim][l]
  __shared__ __align__(16) __bf16 lsP[8][16 * 72];   // per-wave P scratch

  const size_t headBase = ((size_t)b * Hc + h) * (size_t)Sc * KSc;

#if USE_TDM
  if (tid < 32)   // wave 0 issues the Q-tile DMA (128x64, padded rows)
    tdm_load_tile64(lds_addr_of(lsQ), (const void*)(Qb + headBase + (size_t)q0 * KSc), 128);
#else
  {
    const int row = tid >> 1;
    const int cb  = (tid & 1) * 32;
    const uint4* src = reinterpret_cast<const uint4*>(
        Qb + headBase + (size_t)(q0 + row) * KSc + cb);
    *reinterpret_cast<uint4*>(&lsQ[row * 72 + cb])     = src[0];
    *reinterpret_cast<uint4*>(&lsQ[row * 72 + cb + 8]) = src[1];
  }
#endif

  v8f o[4];
  const v8f vzero = {0.f,0.f,0.f,0.f,0.f,0.f,0.f,0.f};
#pragma unroll
  for (int j = 0; j < 4; ++j) o[j] = vzero;
  float mrow[8], lrow[8];
#pragma unroll
  for (int r = 0; r < 8; ++r) { mrow[r] = -3.0e38f; lrow[r] = 0.f; }

  const int wqmax = q0 + 16 * w + 15;
  const int nLt   = (q0 + 128) / 64;

  for (int lt = 0; lt < nLt; ++lt) {
    const int l0 = lt * 64;
    __syncthreads();  // previous-iter readers done before K/V overwrite
#if USE_TDM
    if (tid < 32)     // wave 0 issues the K-tile DMA (64x64, padded rows)
      tdm_load_tile64(lds_addr_of(lsK), (const void*)(Kb + headBase + (size_t)l0 * KSc), 64);
#else
    {
      const int row = tid >> 2;
      const int cb  = (tid & 3) * 16;
      const uint4* sk = reinterpret_cast<const uint4*>(
          Kb + headBase + (size_t)(l0 + row) * KSc + cb);
      *reinterpret_cast<uint4*>(&lsK[row * 72 + cb])     = sk[0];
      *reinterpret_cast<uint4*>(&lsK[row * 72 + cb + 8]) = sk[1];
    }
#endif
    {   // V tile, transposed into [dim][l] (TDM cannot transpose)
      const int row = tid >> 2;
      const int cb  = (tid & 3) * 16;
      const uint4* sv = reinterpret_cast<const uint4*>(
          Vb + headBase + (size_t)(l0 + row) * KSc + cb);
      union { uint4 u[2]; __bf16 e[16]; } t;
      t.u[0] = sv[0]; t.u[1] = sv[1];
#pragma unroll
      for (int i = 0; i < 16; ++i) lsVT[(cb + i) * 72 + row] = t.e[i];
    }
    // warm L2 for the next key tile while this one computes
    if (lt + 1 < nLt && tid < 64) {
      const size_t nb = headBase + (size_t)(l0 + 64) * KSc + (size_t)tid * 64;
      __builtin_prefetch(Kb + nb, 0, 1);
      __builtin_prefetch(Vb + nb, 0, 1);
    }
#if USE_TDM
    if (tid < 32) __builtin_amdgcn_s_wait_tensorcnt(0);
#endif
    __syncthreads();

    if (l0 <= wqmax) {  // wave-uniform: EXEC stays all-1s for WMMA
      // S = Q * K^T  (16 queries x 64 keys)
      v8f sc[4];
#pragma unroll
      for (int j = 0; j < 4; ++j) sc[j] = vzero;
#pragma unroll
      for (int kc = 0; kc < 2; ++kc) {
        const v16bf aF = frag_a(&lsQ[(16 * w + nIdx) * 72 + kc * 32], halfSel);
        v16bf bF[4];
#pragma unroll
        for (int j = 0; j < 4; ++j)
          bF[j] = frag_b(&lsK[(16 * j + nIdx) * 72 + kc * 32], halfSel);
#pragma unroll
        for (int j = 0; j < 4; ++j) sc[j] = WMMA_BF16(aF, bF[j], sc[j]);
      }
      // scale + causal mask + per-lane row max
      float pmax[8];
#pragma unroll
      for (int r = 0; r < 8; ++r) pmax[r] = -3.0e38f;
#pragma unroll
      for (int j = 0; j < 4; ++j)
#pragma unroll
        for (int r = 0; r < 8; ++r) {
          float sv = sc[j][r] * 0.125f;             // 1/sqrt(64)
          const int row = q0 + 16 * w + r + 8 * halfSel;
          const int col = l0 + 16 * j + nIdx;
          if (col > row) sv = -3.0e38f;             // strictly-future mask
          sc[j][r] = sv;
          pmax[r] = fmaxf(pmax[r], sv);
        }
      // xor masks <=8 reduce within each 16-lane half (exactly one row each)
#pragma unroll
      for (int mk = 8; mk >= 1; mk >>= 1)
#pragma unroll
        for (int r = 0; r < 8; ++r)
          pmax[r] = fmaxf(pmax[r], __shfl_xor(pmax[r], mk, 32));

      float rescale[8];
#pragma unroll
      for (int r = 0; r < 8; ++r) {
        const float mn = fmaxf(mrow[r], pmax[r]);
        rescale[r] = __expf(mrow[r] - mn);
        mrow[r] = mn;
      }
      float psum[8];
#pragma unroll
      for (int r = 0; r < 8; ++r) psum[r] = 0.f;
#pragma unroll
      for (int j = 0; j < 4; ++j)
#pragma unroll
        for (int r = 0; r < 8; ++r) {
          const float p = __expf(sc[j][r] - mrow[r]);
          sc[j][r] = p;
          psum[r] += p;
        }
#pragma unroll
      for (int mk = 8; mk >= 1; mk >>= 1)
#pragma unroll
        for (int r = 0; r < 8; ++r) psum[r] += __shfl_xor(psum[r], mk, 32);
#pragma unroll
      for (int r = 0; r < 8; ++r) lrow[r] = lrow[r] * rescale[r] + psum[r];
#pragma unroll
      for (int j = 0; j < 4; ++j)
#pragma unroll
        for (int r = 0; r < 8; ++r) o[j][r] *= rescale[r];

      // spill P (C-layout) to per-wave LDS, reload as A-fragments
#pragma unroll
      for (int j = 0; j < 4; ++j)
#pragma unroll
        for (int r = 0; r < 8; ++r) {
          const int prow = r + 8 * halfSel;
          lsP[w][prow * 72 + 16 * j + nIdx] = (__bf16)sc[j][r];
        }
      // O += P * V   (same-wave LDS RAW: compiler inserts s_wait_dscnt)
#pragma unroll
      for (int kc = 0; kc < 2; ++kc) {
        const v16bf pF = frag_a(&lsP[w][nIdx * 72 + kc * 32], halfSel);
        v16bf vF[4];
#pragma unroll
        for (int j = 0; j < 4; ++j)
          vF[j] = frag_b(&lsVT[(16 * j + nIdx) * 72 + kc * 32], halfSel);
#pragma unroll
        for (int j = 0; j < 4; ++j) o[j] = WMMA_BF16(pF, vF[j], o[j]);
      }
    }
  }

  // normalize and emit heads tile (bf16, [b,h,s,k])
  float rcp[8];
#pragma unroll
  for (int r = 0; r < 8; ++r) rcp[r] = 1.0f / lrow[r];
  const size_t obase = headBase + (size_t)(q0 + 16 * w) * KSc;
#pragma unroll
  for (int j = 0; j < 4; ++j)
#pragma unroll
    for (int r = 0; r < 8; ++r) {
      const int M = r + 8 * halfSel;
      Ob[obase + (size_t)M * KSc + 16 * j + nIdx] = (__bf16)(o[j][r] * rcp[r]);
    }
}

// ---- Kernel 3: heads(reshaped k*H+h) @ kernel -> out ----------------------
__global__ __launch_bounds__(256)
void out_gemm_kernel(const __bf16* __restrict__ Ob,
                     const float* __restrict__ kern,
                     float* __restrict__ out) {
  const int r0   = blockIdx.x * 128;
  const int tid  = threadIdx.x;
  const int w    = tid >> 5;
  const int lane = tid & 31;
  const int nIdx = lane & 15;
  const int halfSel = lane >> 4;

  __shared__ __align__(16) __bf16 lsA[128 * 40];  // heads tile [m][j]
  __shared__ __align__(16) __bf16 lsB[64 * 40];   // kernel tile [n][j]

  v8f acc[4];
  const v8f vzero = {0.f,0.f,0.f,0.f,0.f,0.f,0.f,0.f};
#pragma unroll
  for (int j = 0; j < 4; ++j) acc[j] = vzero;

  for (int j0 = 0; j0 < Hc * KSc; j0 += 32) {
    __syncthreads();
    {
      const int row  = tid >> 1;
      const int jb   = (tid & 1) * 16;
      const int grow = r0 + row;
      const int bb   = grow >> 11;        // /S
      const int ss   = grow & (Sc - 1);
      union { __bf16 e[16]; uint4 u[2]; } pk;
#pragma unroll
      for (int i = 0; i < 16; ++i) {
        const int j  = j0 + jb + i;       // flat feature = k*H + h
        const int hh = j & (Hc - 1);
        const int kk = j >> 4;
        pk.e[i] = Ob[(((size_t)bb * Hc + hh) * Sc + ss) * KSc + kk];
      }
      *reinterpret_cast<uint4*>(&lsA[row * 40 + jb])     = pk.u[0];
      *reinterpret_cast<uint4*>(&lsA[row * 40 + jb + 8]) = pk.u[1];

      const int n   = tid & 63;
      const int jb2 = (tid >> 6) * 8;
      union { __bf16 e[8]; uint4 u; } pw;
#pragma unroll
      for (int i = 0; i < 8; ++i)
        pw.e[i] = (__bf16)kern[(size_t)(j0 + jb2 + i) * KSc + n];
      *reinterpret_cast<uint4*>(&lsB[n * 40 + jb2]) = pw.u;
    }
    __syncthreads();
    const v16bf aF = frag_a(&lsA[(16 * w + nIdx) * 40], halfSel);
    v16bf bF[4];
#pragma unroll
    for (int j = 0; j < 4; ++j)
      bF[j] = frag_b(&lsB[(16 * j + nIdx) * 40], halfSel);
#pragma unroll
    for (int j = 0; j < 4; ++j) acc[j] = WMMA_BF16(aF, bF[j], acc[j]);
  }

  const size_t ob = (size_t)(r0 + 16 * w) * KSc;
#pragma unroll
  for (int j = 0; j < 4; ++j)
#pragma unroll
    for (int r = 0; r < 8; ++r)
      out[ob + (size_t)(r + 8 * halfSel) * KSc + 16 * j + nIdx] = acc[j][r];
}

// ---- host ------------------------------------------------------------------
extern "C" void kernel_launch(void* const* d_in, const int* in_sizes, int n_in,
                              void* d_out, int out_size, void* d_ws, size_t ws_size,
                              hipStream_t stream) {
  (void)in_sizes; (void)n_in; (void)out_size; (void)ws_size;
  const float* x    = (const float*)d_in[0];
  const float* Wq   = (const float*)d_in[1];
  const float* Wk   = (const float*)d_in[2];
  const float* Wv   = (const float*)d_in[3];
  const float* kern = (const float*)d_in[4];

  char* ws = (char*)d_ws;
  const size_t seg = (size_t)Bc * Hc * Sc * KSc * 2;  // 8 MiB per bf16 tensor
  __bf16* Qb = (__bf16*)(ws + 0 * seg);
  __bf16* Kb = (__bf16*)(ws + 1 * seg);
  __bf16* Vb = (__bf16*)(ws + 2 * seg);
  __bf16* Ob = (__bf16*)(ws + 3 * seg);

  dim3 grid(Sc / 128, Hc, Bc);  // 16 x 16 x 2 = 512 blocks
  proj_qkv_kernel<<<grid, 256, 0, stream>>>(x, Wq, Wk, Wv, Qb, Kb, Vb);
  attn_flash_kernel<<<grid, 256, 0, stream>>>(Qb, Kb, Vb, Ob);
  out_gemm_kernel<<<dim3((Bc * Sc) / 128), 256, 0, stream>>>(Ob, kern, (float*)d_out);
}
